// RSSM_82721070121270
// MI455X (gfx1250) — matrix operand order repaired and checked by reading, not compiled, not run on hardware
//
#include <hip/hip_runtime.h>
#include <hip/hip_bf16.h>

// ---------------------------------------------------------------------------
// RSSM rollout for MI455X (gfx1250, wave32, WMMA).
// B=1024 rows are independent -> each workgroup owns 16 batch rows and runs
// all L=256 timesteps locally (h, z carried in LDS). All matmuls use
// v_wmma_f32_16x16x32_bf16 with fp32 accumulation. Weights pre-converted to
// bf16 and transposed to [N][K] in d_ws (L2-resident, reused 256x). obs is
// pre-converted to bf16 and streamed into LDS with global_load_async_to_lds
// (ASYNCcnt), overlapping the copy with the E2/F phases of the prior step.
// ---------------------------------------------------------------------------

#define B_    1024
#define L_    256
#define WD_   128
#define AD_   32
#define DD_   256
#define SD_   64
#define HD_   256
#define OUTW  576   // h(256)|z(64)|pm(64)|ps(64)|qm(64)|qs(64)

typedef __bf16 bf16_t;
typedef __attribute__((ext_vector_type(8)))  bf16_t v8bf;
typedef __attribute__((ext_vector_type(16))) bf16_t v16bf;
typedef __attribute__((ext_vector_type(8)))  float  v8f;

// bf16 workspace layout (element offsets); weights transposed to [N][K]
#define WSO_IN  0                          // [256][96]
#define WSO_IH  (WSO_IN + 256*96)          // [768][256]
#define WSO_HH  (WSO_IH + 768*256)         // [768][256]
#define WSO_P1  (WSO_HH + 768*256)         // [256][256]
#define WSO_P2  (WSO_P1 + 256*256)         // [128][256]
#define WSO_Q1  (WSO_P2 + 128*256)         // [256][384]
#define WSO_Q2  (WSO_Q1 + 256*384)         // [128][256]
#define WSO_OBS (WSO_Q2 + 128*256)         // [B][L][128] bf16 obs

// ---------------- LDS layout (bytes), 63 KB total ---------------------------
#define OFF_HBF   0                        // bf16 [16][256]  persistent h
#define OFF_WBF   (OFF_HBF  + 16*256*2)    // bf16 [16][128]  obs(t) (async dst)
#define OFF_XCAT  (OFF_WBF  + 16*128*2)    // bf16 [16][96]   [z | a_emb]
#define OFF_XBF   (OFF_XCAT + 16*96*2)     // bf16 [16][256]  relu(LN(xin))
#define OFF_GATES (OFF_XBF  + 16*256*2)    // 32 KB scratch region:
                                           //   ru  bf16 [16][512] (sigmoided)
                                           //   gin bf16 [16][256]
                                           //   ghn bf16 [16][256]
                                           // aliased: xin fp32 [16][256] (16KB)
                                           // aliased: pbf/qbf bf16 [16][256]
#define OFF_QSB   (OFF_GATES + 32768)      // fp32 [16][128]  qm|qs
#define LDS_BYTES (OFF_QSB + 16*128*4)     // = 64512 < 65536

// A-fragment (16x32 bf16) from row-major LDS: per ISA 7.12.2 layout,
// dwords0-3 = K[8*half .. 8*half+7], dwords4-7 = K[16+8*half .. +7]
__device__ __forceinline__ v16bf ldsA(const bf16_t* base, int rowStride,
                                      int m, int kbase, int half) {
  const bf16_t* p = base + m * rowStride + kbase + half * 8;
  v8bf lo = *(const v8bf*)(p);
  v8bf hi = *(const v8bf*)(p + 16);
  return __builtin_shufflevector(lo, hi, 0,1,2,3,4,5,6,7,8,9,10,11,12,13,14,15);
}

// B-fragment (32x16 bf16) from transposed bf16 weights [N][K]:
// elems 0-7 = K[8*half .. 8*half+7], elems 8-15 = K[16+8*half .. +7]
__device__ __forceinline__ v16bf glbB(const bf16_t* wt, int Kstride,
                                      int n, int kbase, int half) {
  const bf16_t* p = wt + (size_t)n * Kstride + kbase + half * 8;
  v8bf lo = *(const v8bf*)(p);
  v8bf hi = *(const v8bf*)(p + 16);
  return __builtin_shufflevector(lo, hi, 0,1,2,3,4,5,6,7,8,9,10,11,12,13,14,15);
}

#define WMMA_BF16(a, b, c) \
  __builtin_amdgcn_wmma_f32_16x16x32_bf16(false, (a), false, (b), (short)0, (c), false, false)

// Full-wave (32 lane) xor-butterfly sum via ds_swizzle group-of-32 xor mode:
// offset = xor_mask<<10 | and_mask(0x1f). One DS op per stage, no VALU masks.
__device__ __forceinline__ float waveReduceAdd(float v) {
  v += __int_as_float(__builtin_amdgcn_ds_swizzle(__float_as_int(v), 0x401f)); // ^16
  v += __int_as_float(__builtin_amdgcn_ds_swizzle(__float_as_int(v), 0x201f)); // ^8
  v += __int_as_float(__builtin_amdgcn_ds_swizzle(__float_as_int(v), 0x101f)); // ^4
  v += __int_as_float(__builtin_amdgcn_ds_swizzle(__float_as_int(v), 0x081f)); // ^2
  v += __int_as_float(__builtin_amdgcn_ds_swizzle(__float_as_int(v), 0x041f)); // ^1
  return v;
}

// TRANS-pipe math (v_exp_f32 / v_rcp_f32) instead of branchy libm expansions.
__device__ __forceinline__ float sigmoidf_(float x) {
  return __builtin_amdgcn_rcpf(1.0f + __expf(-x));
}
__device__ __forceinline__ float tanhf_(float x) {
  float e = __expf(-2.0f * __builtin_fabsf(x));
  float r = (1.0f - e) * __builtin_amdgcn_rcpf(1.0f + e);
  return __builtin_copysignf(r, x);
}
__device__ __forceinline__ float softplusf_(float x) {
  return (x > 15.f) ? x : __logf(1.0f + __expf(x));
}

// ------------------------- prep: fp32 -> bf16 (transposed) ------------------
__global__ void wprep_kernel(const float* __restrict__ src, bf16_t* __restrict__ dst,
                             int K, int N) {
  int i = blockIdx.x * blockDim.x + threadIdx.x;
  if (i >= K * N) return;
  int n = i / K;
  int k = i - n * K;
  dst[(size_t)n * K + k] = (bf16_t)src[(size_t)k * N + n];
}

// ------------------------- prep: fp32 -> bf16 (flat) ------------------------
__global__ void cvt_kernel(const float* __restrict__ src, bf16_t* __restrict__ dst,
                           int n) {
  int i = blockIdx.x * blockDim.x + threadIdx.x;
  if (i < n) dst[i] = (bf16_t)src[i];
}

// ------------------------------- main kernel --------------------------------
__global__ __launch_bounds__(512)
void rssm_kernel(const int*   __restrict__ actions,
                 const float* __restrict__ noise,
                 const float* __restrict__ emb,
                 const float* __restrict__ b_in,
                 const float* __restrict__ g_ln,
                 const float* __restrict__ beta_ln,
                 const float* __restrict__ b_ih,
                 const float* __restrict__ b_hh,
                 const float* __restrict__ b_p1,
                 const float* __restrict__ b_p2,
                 const float* __restrict__ b_q1,
                 const float* __restrict__ b_q2,
                 const bf16_t* __restrict__ wsb,
                 float* __restrict__ out) {
  __shared__ __align__(32) unsigned char smem[LDS_BYTES];

  bf16_t* hbf  = (bf16_t*)(smem + OFF_HBF);    // [16][256]
  bf16_t* wbf  = (bf16_t*)(smem + OFF_WBF);    // [16][128]
  bf16_t* xcat = (bf16_t*)(smem + OFF_XCAT);   // [16][96]
  bf16_t* xbf  = (bf16_t*)(smem + OFF_XBF);    // [16][256]
  float*  xin  = (float*)(smem + OFF_GATES);   // [16][256] (aliases gates)
  bf16_t* ru   = (bf16_t*)(smem + OFF_GATES);           // [16][512]
  bf16_t* gin  = (bf16_t*)(smem + OFF_GATES + 16384);   // [16][256]
  bf16_t* ghn  = (bf16_t*)(smem + OFF_GATES + 24576);   // [16][256]
  bf16_t* pbf  = (bf16_t*)(smem + OFF_GATES);           // [16][256] (after D)
  bf16_t* qbf  = (bf16_t*)(smem + OFF_GATES + 8192);    // [16][256]
  float*  qsb  = (float*)(smem + OFF_QSB);     // [16][128]

  const bf16_t* wt_in = wsb + WSO_IN;
  const bf16_t* wt_ih = wsb + WSO_IH;
  const bf16_t* wt_hh = wsb + WSO_HH;
  const bf16_t* wt_p1 = wsb + WSO_P1;
  const bf16_t* wt_p2 = wsb + WSO_P2;
  const bf16_t* wt_q1 = wsb + WSO_Q1;
  const bf16_t* wt_q2 = wsb + WSO_Q2;
  const bf16_t* obsb  = wsb + WSO_OBS;         // bf16 obs [B][L][128]

  const int tid  = threadIdx.x;
  const int wave = tid >> 5;
  const int lane = tid & 31;
  const int l15  = lane & 15;
  const int half = lane >> 4;
  const int b0   = blockIdx.x * 16;

  // ---- hoist loop-invariant per-wave weight fragments into VGPRs ----
  v16bf wBin[3];                               // phase B: wt_in ntile = wave
  {
    int col = 16 * wave + l15;
    for (int kt = 0; kt < 3; ++kt) wBin[kt] = glbB(wt_in, 96, col, 32 * kt, half);
  }
  const float bin_c = b_in[16 * wave + l15];

  v16bf wBhead[8];                             // phase E2: p2 (waves<8) / q2
  float bhead_c;
  {
    int col = (wave < 8) ? (16 * wave + l15) : (16 * (wave - 8) + l15);
    const bf16_t* wt = (wave < 8) ? wt_p2 : wt_q2;
    for (int kt = 0; kt < 8; ++kt) wBhead[kt] = glbB(wt, 256, col, 32 * kt, half);
    bhead_c = (wave < 8) ? b_p2[col] : b_q2[col];
  }

  float gl[8], bl[8];                          // LN gamma/beta for my 8 cols
  for (int j = 0; j < 8; ++j) {
    gl[j] = g_ln[lane + 32 * j];
    bl[j] = beta_ln[lane + 32 * j];
  }

  // ---- init carried state: h = 0, z = 0 ----
  for (int i = tid; i < 16 * 256; i += 512) hbf[i] = (bf16_t)0.0f;
  for (int i = tid; i < 16 * 64;  i += 512) {
    int row = i >> 6, c = i & 63;
    xcat[row * 96 + c] = (bf16_t)0.0f;
  }
  __syncthreads();

  for (int t = 0; t < L_; ++t) {
    // =============== Phase A: a_emb gather; obs(t) into wbf ===============
    if (t == 0) {
      for (int i = tid; i < 16 * 128; i += 512) {
        int r = i >> 7, c = i & 127;
        wbf[r * 128 + c] = obsb[((size_t)(b0 + r) * L_) * WD_ + c];
      }
    } else {
      // obs(t) was issued async last step; wait before anyone reads wbf
      asm volatile("s_wait_asynccnt 0x0" ::: "memory");
    }
    {
      int row = tid >> 5, col = tid & 31;            // 512 = 16x32
      int act = actions[(size_t)(b0 + row) * L_ + t];
      xcat[row * 96 + 64 + col] = (bf16_t)emb[act * AD_ + col];
      if (t + 1 < L_ && (tid & 31) == 0)             // prefetch next noise
        __builtin_prefetch(&noise[((size_t)(b0 + row) * L_ + (t + 1)) * SD_], 0, 1);
    }
    __syncthreads();

    // =============== Phase B: xin = xcat @ w_in + b_in (K=96) =============
    {
      int col = 16 * wave + l15;                     // wave = ntile 0..15
      v8f acc = {};
      for (int kt = 0; kt < 3; ++kt) {
        v16bf a = ldsA(xcat, 96, l15, 32 * kt, half);
        acc = WMMA_BF16(a, wBin[kt], acc);
      }
      for (int r = 0; r < 8; ++r)
        xin[(r + 8 * half) * 256 + col] = acc[r] + bin_c;
    }
    __syncthreads();

    // =============== Phase B2: LayerNorm + ReLU (wave w -> row w) =========
    {
      int row = wave;
      const float* xr = xin + row * 256;
      float v[8], s = 0.f;
      for (int j = 0; j < 8; ++j) { v[j] = xr[lane + 32 * j]; s += v[j]; }
      s = waveReduceAdd(s);
      float mean = s * (1.0f / 256.0f), s2 = 0.f;
      for (int j = 0; j < 8; ++j) { float d = v[j] - mean; s2 += d * d; }
      s2 = waveReduceAdd(s2);
      float rstd = rsqrtf(s2 * (1.0f / 256.0f) + 1e-5f);
      for (int j = 0; j < 8; ++j) {
        int col = lane + 32 * j;
        float y = (v[j] - mean) * rstd * gl[j] + bl[j];
        xbf[row * 256 + col] = (bf16_t)fmaxf(y, 0.f);
      }
    }
    __syncthreads();

    // ====== Phase C: GRU gates. r/u accumulate x@Wih + h@Whh jointly ======
    for (int task = wave; task < 64; task += 16) {
      if (task < 32) {                               // r,u joint: N 0..511
        int col = 16 * task + l15;
        v8f acc = {};
        for (int kt = 0; kt < 8; ++kt) {
          v16bf a = ldsA(xbf, 256, l15, 32 * kt, half);
          v16bf b = glbB(wt_ih, 256, col, 32 * kt, half);
          acc = WMMA_BF16(a, b, acc);
        }
        for (int kt = 0; kt < 8; ++kt) {
          v16bf a = ldsA(hbf, 256, l15, 32 * kt, half);
          v16bf b = glbB(wt_hh, 256, col, 32 * kt, half);
          acc = WMMA_BF16(a, b, acc);
        }
        float bias = b_ih[col] + b_hh[col];
        for (int r = 0; r < 8; ++r)
          ru[(r + 8 * half) * 512 + col] = (bf16_t)sigmoidf_(acc[r] + bias);
      } else if (task < 48) {                        // i_n: Wih cols 512..767
        int j = task - 32, col = 16 * j + l15, colw = 512 + col;
        v8f acc = {};
        for (int kt = 0; kt < 8; ++kt) {
          v16bf a = ldsA(xbf, 256, l15, 32 * kt, half);
          v16bf b = glbB(wt_ih, 256, colw, 32 * kt, half);
          acc = WMMA_BF16(a, b, acc);
        }
        float bias = b_ih[colw];
        for (int r = 0; r < 8; ++r)
          gin[(r + 8 * half) * 256 + col] = (bf16_t)(acc[r] + bias);
      } else {                                       // h_n: Whh cols 512..767
        int j = task - 48, col = 16 * j + l15, colw = 512 + col;
        v8f acc = {};
        for (int kt = 0; kt < 8; ++kt) {
          v16bf a = ldsA(hbf, 256, l15, 32 * kt, half);
          v16bf b = glbB(wt_hh, 256, colw, 32 * kt, half);
          acc = WMMA_BF16(a, b, acc);
        }
        float bias = b_hh[colw];
        for (int r = 0; r < 8; ++r)
          ghn[(r + 8 * half) * 256 + col] = (bf16_t)(acc[r] + bias);
      }
    }
    __syncthreads();

    // ===== Phase D: GRU update; h_new -> hbf (bf16) + out (fp32) ==========
    for (int i = 0; i < 8; ++i) {
      int idx = tid + i * 512;
      int row = idx >> 8, c = idx & 255;
      float r   = (float)ru[row * 512 + c];
      float u   = (float)ru[row * 512 + 256 + c];
      float inn = (float)gin[row * 256 + c];
      float hn  = (float)ghn[row * 256 + c];
      float ho  = (float)hbf[row * 256 + c];
      float n   = tanhf_(inn + r * hn);
      float hv  = (1.f - u) * n + u * ho;
      hbf[row * 256 + c] = (bf16_t)hv;
      out[((size_t)(b0 + row) * L_ + t) * OUTW + c] = hv;
    }
    __syncthreads();

    // ===== Phase E1: p1 on waves 0-7, q1 on waves 8-15 (concurrent) =======
    if (wave < 8) {
      for (int j = wave; j < 16; j += 8) {
        int col = 16 * j + l15;
        v8f acc = {};
        for (int kt = 0; kt < 8; ++kt) {
          v16bf a = ldsA(hbf, 256, l15, 32 * kt, half);
          v16bf b = glbB(wt_p1, 256, col, 32 * kt, half);
          acc = WMMA_BF16(a, b, acc);
        }
        float bias = b_p1[col];
        for (int r = 0; r < 8; ++r)
          pbf[(r + 8 * half) * 256 + col] = (bf16_t)fmaxf(acc[r] + bias, 0.f);
      }
    } else {
      for (int j = wave - 8; j < 16; j += 8) {
        int col = 16 * j + l15;
        v8f acc = {};
        for (int kt = 0; kt < 12; ++kt) {            // K = 256(h) + 128(obs)
          v16bf a = (kt < 8) ? ldsA(hbf, 256, l15, 32 * kt, half)
                             : ldsA(wbf, 128, l15, 32 * (kt - 8), half);
          v16bf b = glbB(wt_q1, 384, col, 32 * kt, half);
          acc = WMMA_BF16(a, b, acc);
        }
        float bias = b_q1[col];
        for (int r = 0; r < 8; ++r)
          qbf[(r + 8 * half) * 256 + col] = (bf16_t)fmaxf(acc[r] + bias, 0.f);
      }
    }
    __syncthreads();

    // ===== Phase E2: issue async obs(t+1) -> wbf, then p2/q2 heads ========
    // wbf's last reader (q1) passed the barrier above; safe to overwrite.
    if (t + 1 < L_ && tid < 256) {                   // 256 x b128 = 4 KB tile
      int row = tid >> 4, ci = tid & 15;
      const bf16_t* g = obsb + ((size_t)(b0 + row) * L_ + (t + 1)) * WD_ + ci * 8;
      unsigned ldsoff = (unsigned)(OFF_WBF + (tid << 4));
      asm volatile("global_load_async_to_lds_b128 %0, %1, off"
                   :: "v"(ldsoff), "v"(g) : "memory");
    }
    if (wave < 8) {
      int col = 16 * wave + l15;
      v8f acc = {};
      for (int kt = 0; kt < 8; ++kt) {
        v16bf a = ldsA(pbf, 256, l15, 32 * kt, half);
        acc = WMMA_BF16(a, wBhead[kt], acc);
      }
      for (int r = 0; r < 8; ++r) {
        int row = r + 8 * half;
        float v = acc[r] + bhead_c;
        int oc;
        if (col < 64) oc = 320 + col;                          // prior mean
        else { v = softplusf_(v) + 0.1f; oc = 384 + (col - 64); }  // prior std
        out[((size_t)(b0 + row) * L_ + t) * OUTW + oc] = v;
      }
    } else {
      int col = 16 * (wave - 8) + l15;
      v8f acc = {};
      for (int kt = 0; kt < 8; ++kt) {
        v16bf a = ldsA(qbf, 256, l15, 32 * kt, half);
        acc = WMMA_BF16(a, wBhead[kt], acc);
      }
      for (int r = 0; r < 8; ++r) {
        int row = r + 8 * half;
        float v = acc[r] + bhead_c;
        int oc;
        if (col < 64) oc = 448 + col;                          // post mean
        else { v = softplusf_(v) + 0.1f; oc = 512 + (col - 64); }  // post std
        qsb[row * 128 + col] = v;
        out[((size_t)(b0 + row) * L_ + t) * OUTW + oc] = v;
      }
    }
    __syncthreads();

    // ===== Phase F: z = qm + qs*eps -> out + next-step xcat[:, :64] =======
    for (int i = 0; i < 2; ++i) {
      int idx = tid + i * 512;                       // 16*64 = 1024 tasks
      int row = idx >> 6, c = idx & 63;
      float eps = noise[((size_t)(b0 + row) * L_ + t) * SD_ + c];
      float qm = qsb[row * 128 + c];
      float qs = qsb[row * 128 + 64 + c];
      float z = qm + qs * eps;
      out[((size_t)(b0 + row) * L_ + t) * OUTW + 256 + c] = z;
      xcat[row * 96 + c] = (bf16_t)z;
    }
    __syncthreads();
  }
}

// ------------------------------- launcher -----------------------------------
extern "C" void kernel_launch(void* const* d_in, const int* in_sizes, int n_in,
                              void* d_out, int out_size, void* d_ws, size_t ws_size,
                              hipStream_t stream) {
  const int*   actions = (const int*)  d_in[0];
  const float* obs     = (const float*)d_in[1];
  const float* noise   = (const float*)d_in[2];
  const float* emb     = (const float*)d_in[3];
  const float* w_in    = (const float*)d_in[4];
  const float* b_in    = (const float*)d_in[5];
  const float* g_ln    = (const float*)d_in[6];
  const float* beta_ln = (const float*)d_in[7];
  const float* w_ih    = (const float*)d_in[8];
  const float* b_ih    = (const float*)d_in[9];
  const float* w_hh    = (const float*)d_in[10];
  const float* b_hh    = (const float*)d_in[11];
  const float* w_p1    = (const float*)d_in[12];
  const float* b_p1    = (const float*)d_in[13];
  const float* w_p2    = (const float*)d_in[14];
  const float* b_p2    = (const float*)d_in[15];
  const float* w_q1    = (const float*)d_in[16];
  const float* b_q1    = (const float*)d_in[17];
  const float* w_q2    = (const float*)d_in[18];
  const float* b_q2    = (const float*)d_in[19];

  bf16_t* wsb = (bf16_t*)d_ws;
  float*  out = (float*)d_out;

  auto prep = [&](const float* src, size_t dstOff, int K, int N) {
    int total = K * N;
    wprep_kernel<<<(total + 255) / 256, 256, 0, stream>>>(src, wsb + dstOff, K, N);
  };
  prep(w_in, WSO_IN,  96, 256);
  prep(w_ih, WSO_IH, 256, 768);
  prep(w_hh, WSO_HH, 256, 768);
  prep(w_p1, WSO_P1, 256, 256);
  prep(w_p2, WSO_P2, 256, 128);
  prep(w_q1, WSO_Q1, 384, 256);
  prep(w_q2, WSO_Q2, 256, 128);

  {
    int total = B_ * L_ * WD_;                    // obs fp32 -> bf16
    cvt_kernel<<<(total + 255) / 256, 256, 0, stream>>>(obs, wsb + WSO_OBS, total);
  }

  rssm_kernel<<<B_ / 16, 512, 0, stream>>>(
      actions, noise, emb, b_in, g_ln, beta_ln,
      b_ih, b_hh, b_p1, b_p2, b_q1, b_q2, wsb, out);
}
